// temporal_30683246363011
// MI455X (gfx1250) — matrix operand order repaired and checked by reading, not compiled, not run on hardware
//
#include <hip/hip_runtime.h>
#include <hip/hip_bf16.h>
#include <cstdint>

// ---------------------------------------------------------------------------
// Fused single-token attention for MI455X (gfx1250, wave32, WMMA).
//   out = Attn(x) with Q from last token, flash-decode style.
// Heavy math (K/V projection, 1.1 TFLOP) runs as bf16 WMMA 16x16x32 with f32
// accumulation; tiny Q/O projections use fp32 WMMA 16x16x4 (M=16 batches).
// K/V are never materialized in HBM: x streams HBM once (~536 MB -> ~23 us
// floor at 23.3 TB/s). LDS staging is double-buffered via
// GLOBAL_LOAD_ASYNC_TO_LDS_B128 + s_wait_asynccnt when available.
// Workspace requirement: ~286 MB.
// ---------------------------------------------------------------------------

typedef __attribute__((ext_vector_type(2)))  float  v2f;
typedef __attribute__((ext_vector_type(8)))  float  v8f;
typedef __attribute__((ext_vector_type(4)))  int    v4i;
typedef __attribute__((ext_vector_type(16))) __bf16 v16bf;

union FragB16 { uint4 u[2]; v16bf v; };

#define BATCH   16
#define SEQ     4096
#define D_MODEL 2048
#define HEADS   16
#define DH      128
#define LDS_PITCH 88   // bf16 elems/row: 176B = 16B-aligned, 44-bank stride -> conflict-free
#define NSTAGES 512    // 16 token blocks * 32 K-chunks

#if defined(__gfx1250__) && __has_builtin(__builtin_amdgcn_global_load_async_to_lds_b128)
#define HAS_ASYNC_LDS 1
#else
#define HAS_ASYNC_LDS 0
#endif

__device__ __forceinline__ void async_cp16(const void* g, void* l) {
#if HAS_ASYNC_LDS
  __builtin_amdgcn_global_load_async_to_lds_b128(
      (__attribute__((address_space(1))) v4i*)(void*)g,
      (__attribute__((address_space(3))) v4i*)l,
      0, 0);
#else
  (void)g; (void)l;
#endif
}

// wait until at most `n` async-LDS loads are outstanding (n is 16 or 0 here)
__device__ __forceinline__ void wait_async_le16() {
#if defined(__gfx1250__)
#if __has_builtin(__builtin_amdgcn_s_wait_asynccnt)
  __builtin_amdgcn_s_wait_asynccnt(16);
#else
  asm volatile("s_wait_asynccnt 0x10" ::: "memory");
#endif
#endif
}
__device__ __forceinline__ void wait_async_le0() {
#if defined(__gfx1250__)
#if __has_builtin(__builtin_amdgcn_s_wait_asynccnt)
  __builtin_amdgcn_s_wait_asynccnt(0);
#else
  asm volatile("s_wait_asynccnt 0x0" ::: "memory");
#endif
#endif
}

// ---------------------------------------------------------------------------
// fp32 -> bf16 bulk convert (vectorized, n multiple of 4)
// ---------------------------------------------------------------------------
__global__ void cvt_f32_to_bf16(const float* __restrict__ src,
                                __bf16* __restrict__ dst, long n) {
  long i = ((long)blockIdx.x * blockDim.x + threadIdx.x) * 4;
  if (i + 3 < n) {
    float4 f = *(const float4*)(src + i);
    __bf16 t[4];
    t[0] = (__bf16)f.x; t[1] = (__bf16)f.y; t[2] = (__bf16)f.z; t[3] = (__bf16)f.w;
    *(uint2*)(dst + i) = *(const uint2*)t;
  }
}

// ---------------------------------------------------------------------------
// out[16 x 2048] = A[16 x 2048] @ W^T + bias, fp32, one wave per 16x16 tile.
// Uses V_WMMA_F32_16X16X4_F32. A rows are strided by ldA.
// A layout (16x4 f32): lane m = lane&15; K = k0 + 2*(lane>>4) + {0,1}.
// B layout (4x16 f32): lane n = lane&15; same K pair (W row n is K-contiguous).
// ---------------------------------------------------------------------------
__global__ void gemm16_rowmajor_wt(const float* __restrict__ A, long ldA,
                                   const float* __restrict__ W,
                                   const float* __restrict__ bias,
                                   float* __restrict__ out) {
  const int lane = threadIdx.x & 31;
  const int nlo  = lane & 15;
  const int hl   = lane >> 4;
  const int n0   = blockIdx.x * 16;
  v8f acc = {0.f, 0.f, 0.f, 0.f, 0.f, 0.f, 0.f, 0.f};

#if __has_builtin(__builtin_amdgcn_wmma_f32_16x16x4_f32)
  const float* arow = A + (long)nlo * ldA;
  const float* wrow = W + (long)(n0 + nlo) * D_MODEL;
  for (int k0 = 0; k0 < D_MODEL; k0 += 4) {
    const int kk = k0 + 2 * hl;
    v2f av = *(const v2f*)(arow + kk);
    v2f bv = *(const v2f*)(wrow + kk);
    acc = __builtin_amdgcn_wmma_f32_16x16x4_f32(false, av, false, bv,
                                                (short)0, acc, false, false);
  }
  #pragma unroll
  for (int r = 0; r < 8; ++r) {
    const int m = r + 8 * hl;
    out[(long)m * D_MODEL + n0 + nlo] = acc[r] + bias[n0 + nlo];
  }
#else
  #pragma unroll
  for (int r = 0; r < 8; ++r) {
    const int m = r + 8 * hl;
    float s = 0.f;
    for (int k = 0; k < D_MODEL; ++k)
      s += A[(long)m * ldA + k] * W[(long)(n0 + nlo) * D_MODEL + k];
    out[(long)m * D_MODEL + n0 + nlo] = s + bias[n0 + nlo];
  }
#endif
}

// ---------------------------------------------------------------------------
// Fused K/V projection + flash decode. One WG per (batch, head), 256 threads
// (8 waves). 16 blocks of 256 tokens; per block, 32 K-chunks of 64 staged
// into double-buffered LDS (async-to-LDS when available). Each wave owns
// 2 token-strips x 128 head-cols of both k and v accumulators
// (2x2x8 = 32 bf16 WMMAs per 32-wide K step).
// ---------------------------------------------------------------------------
__global__ void __launch_bounds__(256) attn_fused(
    const __bf16* __restrict__ xbf,    // [16][4096][2048]
    const __bf16* __restrict__ wkbf,   // [2048][2048]
    const __bf16* __restrict__ wvbf,   // [2048][2048]
    const float*  __restrict__ qv,     // [16][2048] fp32 (Q of last token)
    const float*  __restrict__ bv,     // [2048]
    float*        __restrict__ inter)  // [16][2048] fp32
{
  __shared__ __bf16 xs [2][256 * LDS_PITCH];  // 2 x 44 KB
  __shared__ __bf16 wks[2][128 * LDS_PITCH];  // 2 x 22 KB
  __shared__ __bf16 wvs[2][128 * LDS_PITCH];  // 2 x 22 KB
  __shared__ float s_lds[256];
  __shared__ float w_lds[256];
  __shared__ float red[16];
  __shared__ float o_acc[128];
  __shared__ float mst[4];                    // 0:m  1:sumexp  2:alpha

  const int tid  = threadIdx.x;
  const int lane = tid & 31;
  const int wave = tid >> 5;
  const int nlo  = lane & 15;
  const int hl   = lane >> 4;
  const int b    = blockIdx.x >> 4;
  const int h    = blockIdx.x & 15;

  const long xbase = (long)b * SEQ * D_MODEL;
  const long wbase = (long)h * DH * D_MODEL;

  // q values for this lane's column set (constant over the whole kernel)
  float qreg[8];
  #pragma unroll
  for (int t = 0; t < 8; ++t)
    qreg[t] = qv[(long)b * D_MODEL + h * DH + t * 16 + nlo];

  if (tid == 0) { mst[0] = -__builtin_inff(); mst[1] = 0.f; }
  if (tid < 128) o_acc[tid] = 0.f;

  // -------- stage issue helper (16 x b128 per thread per stage) -----------
  auto issue_stage = [&](int gst, int buf) {
    const int T0 = (gst >> 5) * 256;
    const int k0 = (gst & 31) * 64;
    #pragma unroll
    for (int i = 0; i < 8; ++i) {
      const int idx = tid + i * 256;
      const int tok = idx >> 3, kk = (idx & 7) * 8;
      async_cp16(xbf + xbase + (long)(T0 + tok) * D_MODEL + k0 + kk,
                 &xs[buf][tok * LDS_PITCH + kk]);
    }
    #pragma unroll
    for (int i = 0; i < 4; ++i) {
      const int idx = tid + i * 256;
      const int row = idx >> 3, kk = (idx & 7) * 8;
      async_cp16(wkbf + wbase + (long)row * D_MODEL + k0 + kk,
                 &wks[buf][row * LDS_PITCH + kk]);
      async_cp16(wvbf + wbase + (long)row * D_MODEL + k0 + kk,
                 &wvs[buf][row * LDS_PITCH + kk]);
    }
  };

  // synchronous staging fallback
  auto stage_sync = [&](int gst, int buf) {
    const int T0 = (gst >> 5) * 256;
    const int k0 = (gst & 31) * 64;
    #pragma unroll
    for (int i = 0; i < 8; ++i) {
      const int idx = tid + i * 256;
      const int tok = idx >> 3, kk = (idx & 7) * 8;
      *(uint4*)&xs[buf][tok * LDS_PITCH + kk] =
          *(const uint4*)(xbf + xbase + (long)(T0 + tok) * D_MODEL + k0 + kk);
    }
    #pragma unroll
    for (int i = 0; i < 4; ++i) {
      const int idx = tid + i * 256;
      const int row = idx >> 3, kk = (idx & 7) * 8;
      *(uint4*)&wks[buf][row * LDS_PITCH + kk] =
          *(const uint4*)(wkbf + wbase + (long)row * D_MODEL + k0 + kk);
      *(uint4*)&wvs[buf][row * LDS_PITCH + kk] =
          *(const uint4*)(wvbf + wbase + (long)row * D_MODEL + k0 + kk);
    }
  };

  const v8f vz = {0.f, 0.f, 0.f, 0.f, 0.f, 0.f, 0.f, 0.f};

#if HAS_ASYNC_LDS
  issue_stage(0, 0);            // prologue prefetch
#endif

  int gst = 0;
  for (int bi = 0; bi < 16; ++bi) {        // 16 token blocks of 256
    v8f ak[2][8], avv[2][8];
    #pragma unroll
    for (int s = 0; s < 2; ++s)
      #pragma unroll
      for (int t = 0; t < 8; ++t) { ak[s][t] = vz; avv[s][t] = vz; }

    for (int st = 0; st < 32; ++st, ++gst) {   // 32 K-chunks of 64
      const int cb = gst & 1;
      __syncthreads();                     // barrier A: next buffer free
#if HAS_ASYNC_LDS
      if (gst + 1 < NSTAGES) {
        issue_stage(gst + 1, cb ^ 1);      // prefetch next chunk
        wait_async_le16();                 // current buffer resident
      } else {
        wait_async_le0();
      }
#else
      stage_sync(gst, cb);
#endif
      __syncthreads();                     // barrier B: all waves see buffer

      #pragma unroll
      for (int ks = 0; ks < 2; ++ks) {     // two 32-wide WMMA K steps
        const int kb = ks * 32;
        // A fragments (16x32 bf16): lanes<16 hold K runs [0..8)+[16..24),
        // lanes>=16 hold [8..16)+[24..32)
        FragB16 a[2];
        #pragma unroll
        for (int s = 0; s < 2; ++s) {
          const __bf16* p =
              &xs[cb][((wave * 2 + s) * 16 + nlo) * LDS_PITCH + kb + 8 * hl];
          a[s].u[0] = *(const uint4*)p;
          a[s].u[1] = *(const uint4*)(p + 16);
        }
        #pragma unroll
        for (int t = 0; t < 8; ++t) {
          // B fragments (32x16 bf16): lanes<16 hold K [0..8)+[16..24), lanes>=16 [8..16)+[24..32)
          FragB16 bk_, bv_;
          const int row = t * 16 + nlo;
          const __bf16* pk = &wks[cb][row * LDS_PITCH + kb + 8 * hl];
          bk_.u[0] = *(const uint4*)pk; bk_.u[1] = *(const uint4*)(pk + 16);
          const __bf16* pv = &wvs[cb][row * LDS_PITCH + kb + 8 * hl];
          bv_.u[0] = *(const uint4*)pv; bv_.u[1] = *(const uint4*)(pv + 16);
          #pragma unroll
          for (int s = 0; s < 2; ++s) {
            ak[s][t]  = __builtin_amdgcn_wmma_f32_16x16x32_bf16(
                false, a[s].v, false, bk_.v, (short)0, ak[s][t], false, false);
            avv[s][t] = __builtin_amdgcn_wmma_f32_16x16x32_bf16(
                false, a[s].v, false, bv_.v, (short)0, avv[s][t], false, false);
          }
        }
      }
    }

    // scores: s[t] = (q . k[t]) / sqrt(128); C layout: col=nlo, token=r+8*hl
    #pragma unroll
    for (int s = 0; s < 2; ++s) {
      const int strip = wave * 2 + s;
      #pragma unroll
      for (int r = 0; r < 8; ++r) {
        float p = 0.f;
        #pragma unroll
        for (int t = 0; t < 8; ++t) p += qreg[t] * ak[s][t][r];
        p += __shfl_xor(p, 1, 32);
        p += __shfl_xor(p, 2, 32);
        p += __shfl_xor(p, 4, 32);
        p += __shfl_xor(p, 8, 32);
        if (nlo == 0) s_lds[strip * 16 + hl * 8 + r] = p * 0.08838834764831845f;
      }
    }
    __syncthreads();

    // block max: wave-level butterflies + 8-entry LDS handoff
    {
      float v = s_lds[tid];
      #pragma unroll
      for (int m = 16; m >= 1; m >>= 1) v = fmaxf(v, __shfl_xor(v, m, 32));
      if (lane == 0) red[wave] = v;
    }
    __syncthreads();
    if (tid == 0) {
      float mb = red[0];
      #pragma unroll
      for (int i = 1; i < 8; ++i) mb = fmaxf(mb, red[i]);
      const float mo = mst[0];
      const float mn2 = fmaxf(mo, mb);
      mst[2] = __expf(mo - mn2);
      mst[0] = mn2;
    }
    __syncthreads();
    const float mn = mst[0], alpha = mst[2];
    {
      const float w = __expf(s_lds[tid] - mn);
      w_lds[tid] = w;
      float sw = w;
      #pragma unroll
      for (int m = 16; m >= 1; m >>= 1) sw += __shfl_xor(sw, m, 32);
      if (lane == 0) red[8 + wave] = sw;
    }
    if (tid < 128) o_acc[tid] *= alpha;
    __syncthreads();
    if (tid == 0) {
      float s8 = 0.f;
      #pragma unroll
      for (int i = 0; i < 8; ++i) s8 += red[8 + i];
      mst[1] = mst[1] * alpha + s8;
    }

    // o_acc[c] += sum_t w[t] * v[t, c]
    #pragma unroll
    for (int s = 0; s < 2; ++s) {
      const int strip = wave * 2 + s;
      #pragma unroll
      for (int t = 0; t < 8; ++t) {
        float c = 0.f;
        #pragma unroll
        for (int r = 0; r < 8; ++r)
          c += w_lds[strip * 16 + hl * 8 + r] * avv[s][t][r];
        c += __shfl_xor(c, 16, 32);
        if (hl == 0) atomicAdd(&o_acc[t * 16 + nlo], c);
      }
    }
    // next stage's barrier A orders atomics / LDS reuse
  }
  __syncthreads();
  if (tid < 128)
    inter[(long)b * D_MODEL + h * DH + tid] = o_acc[tid] / mst[1] + bv[h * DH + tid];
}

// ---------------------------------------------------------------------------
extern "C" void kernel_launch(void* const* d_in, const int* in_sizes, int n_in,
                              void* d_out, int out_size, void* d_ws, size_t ws_size,
                              hipStream_t stream) {
  const float* x  = (const float*)d_in[0];
  const float* Wq = (const float*)d_in[1];
  const float* bq = (const float*)d_in[2];
  const float* Wk = (const float*)d_in[3];
  // d_in[4] = bk: adds q.bk (constant across tokens) to every score of a
  // (b,h) pair -> exactly cancels in softmax; intentionally skipped.
  const float* Wv = (const float*)d_in[5];
  const float* bv = (const float*)d_in[6];
  const float* Wo = (const float*)d_in[7];
  const float* bo = (const float*)d_in[8];

  char* p = (char*)d_ws;
  __bf16* xbf  = (__bf16*)p; p += (size_t)BATCH * SEQ * D_MODEL * 2;   // 256 MB
  __bf16* wkbf = (__bf16*)p; p += (size_t)D_MODEL * D_MODEL * 2;       //   8 MB
  __bf16* wvbf = (__bf16*)p; p += (size_t)D_MODEL * D_MODEL * 2;       //   8 MB
  float*  qws  = (float*)p;  p += (size_t)BATCH * D_MODEL * 4;
  float*  intr = (float*)p;  p += (size_t)BATCH * D_MODEL * 4;

  const long nx = (long)BATCH * SEQ * D_MODEL;       // 134217728
  const long nw = (long)D_MODEL * D_MODEL;           //   4194304

  cvt_f32_to_bf16<<<(int)(nx / 1024), 256, 0, stream>>>(x,  xbf,  nx);
  cvt_f32_to_bf16<<<(int)(nw / 1024), 256, 0, stream>>>(Wk, wkbf, nw);
  cvt_f32_to_bf16<<<(int)(nw / 1024), 256, 0, stream>>>(Wv, wvbf, nw);

  // Q projection from last tokens (fp32 WMMA), rows strided by SEQ*D_MODEL
  gemm16_rowmajor_wt<<<D_MODEL / 16, 32, 0, stream>>>(
      x + (long)(SEQ - 1) * D_MODEL, (long)SEQ * D_MODEL, Wq, bq, qws);

  attn_fused<<<BATCH * HEADS, 256, 0, stream>>>(xbf, wkbf, wvbf, qws, bv, intr);

  // output projection
  gemm16_rowmajor_wt<<<D_MODEL / 16, 32, 0, stream>>>(
      intr, (long)D_MODEL, Wo, bo, (float*)d_out);
}
